// V2FConv3d_10763188043851
// MI455X (gfx1250) — compile-verified, matrix operand order checked by reading, not run on hardware
//
#include <hip/hip_runtime.h>
#include <hip/hip_bf16.h>
#include <stdint.h>

typedef __attribute__((ext_vector_type(16))) _Float16 v16h;
typedef __attribute__((ext_vector_type(8)))  float    v8f;
typedef __attribute__((ext_vector_type(4)))  float    v4f;

#define C_IN   64
#define C_OUT  64
#define BN_EPSF 1e-5f

// ---------------------------------------------------------------------------
// pass0: pre-swizzle depth_weights[64,64] f32 -> f16 WMMA B-fragments in ws,
//        and zero the BN stat accumulators (ws is poisoned by the harness).
// B-fragment layout (CDNA5 16-bit operand layout, wave32):
//   lane L: col n = 16*t + (L&15); half-select base = 8*(L>>4)
//   half h (0..15): K = frag*32 + (h<8 ? base+h : 16+base+(h-8))
// stored as bfrag[((t*2+frag)*32 + lane)*16 + h]
// ---------------------------------------------------------------------------
__global__ void v2f_pass0(const float* __restrict__ dw, _Float16* __restrict__ bfrag,
                          float* __restrict__ gsum, float* __restrict__ gsumsq)
{
    int tid = threadIdx.x;                    // 256 threads
    if (tid < C_OUT) { gsum[tid] = 0.f; gsumsq[tid] = 0.f; }
    int t    = tid >> 6;                      // 16-col output tile 0..3
    int frag = (tid >> 5) & 1;                // K half: 0 -> K 0..31, 1 -> K 32..63
    int lane = tid & 31;
    int base = (lane >> 4) * 8;
    int n    = t * 16 + (lane & 15);
    _Float16* dst = bfrag + (size_t)((t * 2 + frag) * 32 + lane) * 16;
#pragma unroll
    for (int h = 0; h < 16; ++h) {
        int k = frag * 32 + ((h < 8) ? (base + h) : (16 + base + (h - 8)));
        dst[h] = (_Float16)dw[k * C_OUT + n];
    }
}

// ---------------------------------------------------------------------------
// pass1: gather + depthwise spatial conv + 64x64 WMMA GEMM + bias + ReLU,
//        write pre-BN output, accumulate per-channel sum / sumsq.
// Block = 128 threads = 4 wave32s; each wave owns 16 facets (16x64 out tile).
// Epilogue: wave-uniform full-tile test, stores via one base pointer +
// constant immediate offsets (element (t,e) lives at po[e*64 + t*16]).
// ---------------------------------------------------------------------------
__global__ void __launch_bounds__(128)
v2f_pass1(const float* __restrict__ inputs, const float* __restrict__ sw,
          const int* __restrict__ face, const _Float16* __restrict__ bfrag,
          const float* __restrict__ bias, float* __restrict__ out,
          float* __restrict__ gsum, float* __restrict__ gsumsq, int F)
{
    __shared__ float lds_sum[C_OUT];
    __shared__ float lds_sq[C_OUT];
    int tid = threadIdx.x;
    if (tid < C_OUT) { lds_sum[tid] = 0.f; lds_sq[tid] = 0.f; }
    __syncthreads();

    int wave = tid >> 5;           // wave32
    int lane = tid & 31;
    int hs   = lane >> 4;          // half-wave select
    int mrow = lane & 15;          // A-matrix row within tile
    int base = hs * 8;             // K-offset for this half-wave

    int64_t fbase = (int64_t)blockIdx.x * 64 + (int64_t)wave * 16;
    int64_t f = fbase + mrow;
    if (f > (int64_t)F - 1) f = (int64_t)F - 1;   // clamp (F%64==0 anyway)

    int i0 = face[f * 3 + 0];
    int i1 = face[f * 3 + 1];
    int i2 = face[f * 3 + 2];

    // v2f[mrow, c] for the 32 channels this lane owns (4 runs of 8, matching
    // the 16-bit A-fragment layout): run r covers c = base + 16*r + 0..7
    float acc[32];
#pragma unroll
    for (int j = 0; j < 32; ++j) acc[j] = 0.f;

    auto accum = [&](const float* __restrict__ row, const float* __restrict__ swk) {
#pragma unroll
        for (int r = 0; r < 4; ++r) {
            int c0 = base + 16 * r;
            v4f a0 = *(const v4f*)(row + c0);
            v4f a1 = *(const v4f*)(row + c0 + 4);
            v4f w0 = *(const v4f*)(swk + c0);
            v4f w1 = *(const v4f*)(swk + c0 + 4);
#pragma unroll
            for (int j = 0; j < 4; ++j) {
                acc[r * 8 + j]     += a0[j] * w0[j];
                acc[r * 8 + 4 + j] += a1[j] * w1[j];
            }
        }
    };
    accum(inputs + (size_t)i0 * C_IN, sw);
    accum(inputs + (size_t)i1 * C_IN, sw + C_IN);
    accum(inputs + (size_t)i2 * C_IN, sw + 2 * C_IN);

    // Pack A fragments: alo covers K=0..31, ahi covers K=32..63
    v16h alo, ahi;
#pragma unroll
    for (int h = 0; h < 16; ++h) {
        alo[h] = (_Float16)acc[h];
        ahi[h] = (_Float16)acc[16 + h];
    }

    // All 4 output tiles: 8 WMMAs, B-fragment loads issued up front.
    v8f cacc[4];
#pragma unroll
    for (int t = 0; t < 4; ++t) {
        v16h blo = *(const v16h*)(bfrag + (size_t)((t * 2 + 0) * 32 + lane) * 16);
        v16h bhi = *(const v16h*)(bfrag + (size_t)((t * 2 + 1) * 32 + lane) * 16);
        v8f c = {};
        c = __builtin_amdgcn_wmma_f32_16x16x32_f16(false, alo, false, blo,
                                                   (short)0, c, false, false);
        cacc[t] = __builtin_amdgcn_wmma_f32_16x16x32_f16(false, ahi, false, bhi,
                                                         (short)0, c, false, false);
    }

    // Bias per tile column (C/D layout: N = mrow + 16t, M = e + 8*hs)
    float bb[4];
#pragma unroll
    for (int t = 0; t < 4; ++t) bb[t] = bias[t * 16 + mrow];

    // One base pointer; element (t,e) at constant offset e*64 + t*16.
    float* po = out + (fbase + 8 * hs) * C_OUT + mrow;

    float s[4]  = {0.f, 0.f, 0.f, 0.f};
    float s2[4] = {0.f, 0.f, 0.f, 0.f};

    if (fbase + 16 <= (int64_t)F) {            // wave-uniform fast path
#pragma unroll
        for (int e = 0; e < 8; ++e) {
#pragma unroll
            for (int t = 0; t < 4; ++t) {
                float v = fmaxf(cacc[t][e] + bb[t], 0.f);
                po[e * C_OUT + t * 16] = v;    // immediate-offset store
                s[t]  += v;
                s2[t] += v * v;
            }
        }
    } else {                                   // tail tile (unused when F%64==0)
#pragma unroll
        for (int e = 0; e < 8; ++e) {
            int64_t row = fbase + 8 * hs + e;
            if (row < (int64_t)F) {
#pragma unroll
                for (int t = 0; t < 4; ++t) {
                    float v = fmaxf(cacc[t][e] + bb[t], 0.f);
                    po[e * C_OUT + t * 16] = v;
                    s[t]  += v;
                    s2[t] += v * v;
                }
            }
        }
    }

#pragma unroll
    for (int t = 0; t < 4; ++t) {
        int ncol = t * 16 + mrow;
        atomicAdd(&lds_sum[ncol], s[t]);       // ds_add_f32
        atomicAdd(&lds_sq[ncol],  s2[t]);
    }
    __syncthreads();
    if (tid < C_OUT) {
        atomicAdd(&gsum[tid],   lds_sum[tid]); // one global atomic per channel per block
        atomicAdd(&gsumsq[tid], lds_sq[tid]);
    }
}

// ---------------------------------------------------------------------------
// pass2: in-place BatchNorm normalize (biased variance, matching jnp.var)
// ---------------------------------------------------------------------------
__global__ void v2f_pass2(float* __restrict__ out, const float* __restrict__ gsum,
                          const float* __restrict__ gsumsq, const float* __restrict__ gamma,
                          const float* __restrict__ beta, int64_t total, float invF)
{
    int64_t i = ((int64_t)blockIdx.x * blockDim.x + threadIdx.x) * 4;
    if (i >= total) return;
    v4f v = *(v4f*)(out + i);
    int c0 = (int)(i & (C_OUT - 1));
#pragma unroll
    for (int j = 0; j < 4; ++j) {
        int c = c0 + j;
        float mean = gsum[c] * invF;
        float var  = gsumsq[c] * invF - mean * mean;
        float rs   = rsqrtf(var + BN_EPSF);
        v[j] = (v[j] - mean) * rs * gamma[c] + beta[c];
    }
    *(v4f*)(out + i) = v;
}

// ---------------------------------------------------------------------------
extern "C" void kernel_launch(void* const* d_in, const int* in_sizes, int n_in,
                              void* d_out, int out_size, void* d_ws, size_t ws_size,
                              hipStream_t stream)
{
    const float* inputs = (const float*)d_in[0];   // [V, 64]
    const float* sw     = (const float*)d_in[1];   // [3, 64, 1]
    const float* dw     = (const float*)d_in[2];   // [64, 64]
    const float* bias   = (const float*)d_in[3];   // [1, 64]
    const float* gamma  = (const float*)d_in[4];   // [64]
    const float* beta   = (const float*)d_in[5];   // [64]
    const int*   face   = (const int*)d_in[6];     // [F, 3]
    int F = in_sizes[6] / 3;
    float* out = (float*)d_out;                    // [F, 64]

    _Float16* bfrag  = (_Float16*)d_ws;            // 4*2*32*16 halves = 8 KB
    float*    gsum   = (float*)((char*)d_ws + 8192);
    float*    gsumsq = gsum + C_OUT;

    v2f_pass0<<<1, 256, 0, stream>>>(dw, bfrag, gsum, gsumsq);

    int nblk1 = (F + 63) / 64;                     // 15625 for F=1M (exact)
    v2f_pass1<<<nblk1, 128, 0, stream>>>(inputs, sw, face, bfrag, bias,
                                         out, gsum, gsumsq, F);

    int64_t total = (int64_t)out_size;             // F*64, multiple of 4
    int nblk2 = (int)((total / 4 + 255) / 256);
    v2f_pass2<<<nblk2, 256, 0, stream>>>(out, gsum, gsumsq, gamma, beta,
                                         total, 1.0f / (float)F);
}
